// HNN_44736379355163
// MI455X (gfx1250) — matrix operand order repaired and checked by reading, not compiled, not run on hardware
//
#include <hip/hip_runtime.h>
#include <hip/hip_bf16.h>
#include <cstddef>

typedef __attribute__((ext_vector_type(16))) _Float16 v16h;
typedef __attribute__((ext_vector_type(8)))  _Float16 v8h;
typedef __attribute__((ext_vector_type(8)))  float    v8f;
typedef __attribute__((ext_vector_type(4)))  float    v4f;

static __device__ __forceinline__ v8f wmma16(v16h a, v16h b, v8f c) {
  return __builtin_amdgcn_wmma_f32_16x16x32_f16(false, a, false, b, (short)0, c, false, false);
}

// A-operand lane fragment: one contiguous 16-f16 run at kb+lhalf*16
static __device__ __forceinline__ v16h loadA(const _Float16* __restrict__ row, int kb, int lhalf) {
  return *(const v16h*)(row + kb + lhalf * 16);
}

// ---------------- conversion / packing ----------------

__global__ void cvt_f32_to_f16(const float* __restrict__ src, _Float16* __restrict__ dst, size_t n) {
  size_t i = (size_t)blockIdx.x * blockDim.x + threadIdx.x;
  if (i < n) dst[i] = (_Float16)src[i];
}

// x [256][90][32][32] f32 (NCHW) -> padded NHWC f16 [256][34][34][96]
__global__ void pad_input_x(const float* __restrict__ x, _Float16* __restrict__ xhp) {
  int idx = blockIdx.x * blockDim.x + threadIdx.x;
  const int n = 256 * 34 * 34 * 96;
  if (idx >= n) return;
  int c    = idx % 96;
  int rest = idx / 96;
  int xx   = rest % 34;
  int r2   = rest / 34;
  int yy   = r2 % 34;
  int b    = r2 / 34;
  float v = 0.0f;
  if (c < 90 && yy >= 1 && yy <= 32 && xx >= 1 && xx <= 32)
    v = x[((size_t)(b * 90 + c) * 32 + (yy - 1)) * 32 + (xx - 1)];
  xhp[idx] = (_Float16)v;
}

// encoder weights [32][Cin][3][3] -> f16 [32][Ksteps*32]; chunk ks: tap=ks/cpt, ic=(ks%cpt)*32+slot
__global__ void pack_enc_w(const float* __restrict__ w, _Float16* __restrict__ wpk,
                           int Cin, int cpt, int Ksteps) {
  int Kw = Ksteps * 32;
  int n = 32 * Kw;
  int idx = blockIdx.x * blockDim.x + threadIdx.x;
  if (idx >= n) return;
  int m  = idx / Kw;
  int k  = idx - m * Kw;
  int ks = k >> 5;
  int s  = k & 31;
  int tap = ks / cpt;
  int ic  = (ks - tap * cpt) * 32 + s;
  wpk[idx] = (ic < Cin) ? (_Float16)w[((size_t)m * Cin + ic) * 9 + tap] : (_Float16)0.0f;
}

// transposed-conv weights w[16][Cout][5][5] -> f16 A[16][416]; tap=ks*2+slot/16, ic=slot%16
__global__ void pack_dec_w(const float* __restrict__ w, _Float16* __restrict__ wpk, int Cout) {
  int idx = blockIdx.x * blockDim.x + threadIdx.x;
  const int n = 16 * 416;
  if (idx >= n) return;
  int oc = idx / 416;
  int k  = idx - oc * 416;
  int ks = k >> 5;
  int s  = k & 31;
  int tap = ks * 2 + (s >> 4);
  int ic  = s & 15;
  _Float16 v = (_Float16)0.0f;
  if (oc < Cout && tap < 25) {
    int ky = tap / 5, kx = tap - (tap / 5) * 5;
    v = (_Float16)w[(((size_t)ic * Cout + oc) * 5 + (4 - ky)) * 5 + (4 - kx)];
  }
  wpk[idx] = v;
}

// NHWC maxpool2+relu, C=32; writes zero-halo padded map (off=1) or unpadded (off=0)
__global__ void maxpool_pad_nhwc(const _Float16* __restrict__ pre, _Float16* __restrict__ out,
                                 int B, int H, int W, int Hp, int Wp, int off) {
  int n = B * Hp * Wp * 32;
  int idx = blockIdx.x * blockDim.x + threadIdx.x;
  if (idx >= n) return;
  int c    = idx & 31;
  int rest = idx >> 5;
  int xx   = rest % Wp;
  int r2   = rest / Wp;
  int yy   = r2 % Hp;
  int b    = r2 / Hp;
  int yo = yy - off, xo = xx - off;
  float m = 0.0f;
  if (yo >= 0 && yo < (H >> 1) && xo >= 0 && xo < (W >> 1)) {
    const _Float16* p = pre + (((size_t)b * H + 2 * yo) * W + 2 * xo) * 32 + c;
    m = (float)p[0];
    m = fmaxf(m, (float)p[32]);
    m = fmaxf(m, (float)p[W * 32]);
    m = fmaxf(m, (float)p[W * 32 + 32]);
    m = fmaxf(m, 0.0f);
  }
  out[idx] = (_Float16)m;
}

// last pool: NHWC [256][8][8][32] -> NCHW-flattened h3 [256][512] (feature = c*16+y*4+x)
__global__ void maxpool3_to_h3(const _Float16* __restrict__ pre, _Float16* __restrict__ h3) {
  int idx = blockIdx.x * blockDim.x + threadIdx.x;
  if (idx >= 256 * 512) return;
  int b  = idx >> 9;
  int f  = idx & 511;
  int c  = f >> 4;
  int yo = (f >> 2) & 3;
  int xo = f & 3;
  const _Float16* p = pre + (((size_t)b * 8 + 2 * yo) * 8 + 2 * xo) * 32 + c;
  float m = (float)p[0];
  m = fmaxf(m, (float)p[32]);
  m = fmaxf(m, (float)p[256]);
  m = fmaxf(m, (float)p[288]);
  h3[idx] = (_Float16)fmaxf(m, 0.0f);
}

// z = mu + exp(0.5*logvar)*eps ; qs stored NHWC [t*256+b][4][4][16]
__global__ void qs_kernel(const float* __restrict__ mu, const float* __restrict__ logvar,
                          const float* __restrict__ eps, const float* __restrict__ hamw,
                          _Float16* __restrict__ qsh) {
  int idx = blockIdx.x * blockDim.x + threadIdx.x;
  const int n = 30 * 256 * 256;
  if (idx >= n) return;
  int j = idx & 255;              // feature = ic*16 + iy*4 + ix
  int b = (idx >> 8) & 255;
  int t = idx >> 16;
  int bi = b * 512 + j;
  float z = mu[bi] + expf(0.5f * logvar[bi]) * eps[bi];
  float q = z + (float)t * (0.125f * hamw[256 + j]);
  int nhwc = (idx & ~255) + ((j & 15) << 4) + (j >> 4);   // [spatial][channel]
  qsh[nhwc] = (_Float16)q;
}

// ---------------- encoder conv: NHWC implicit GEMM ----------------
// in: padded NHWC [B][Hh][Wh][Cinp]; out: NHWC [B][H][W][32].
// Wave: C[32 couts][32 positions]; chunk ks -> tap=ks/cpt (dy,dx), icb=(ks%cpt)*32.
__global__ __launch_bounds__(256) void conv_enc_wmma(
    const _Float16* __restrict__ in, const _Float16* __restrict__ wpk,
    const float* __restrict__ bias, _Float16* __restrict__ outpre,
    int Hh, int Wh, int H, int W, int Ksteps, int cpt, int Kstride, int Cinp) {
  __shared__ __align__(64) _Float16 wlds[32 * 864];
  const int tid = threadIdx.x;
  const int lane = tid & 31, lhalf = lane >> 4, lm = lane & 15;
  for (int i = tid; i < 32 * Kstride; i += 256) wlds[i] = wpk[i];
  __syncthreads();

  const int wid = blockIdx.x * 8 + (tid >> 5);
  const int HW = H * W;
  const size_t imgStride = (size_t)Hh * Wh * Cinp;
  int pos0 = wid * 32 + lm, pos1 = pos0 + 16;
  int b0 = pos0 / HW, r0 = pos0 - b0 * HW, y0 = r0 / W, x0 = r0 - y0 * W;
  int b1 = pos1 / HW, r1 = pos1 - b1 * HW, y1 = r1 / W, x1 = r1 - y1 * W;
  const _Float16* img0 = in + (size_t)b0 * imgStride + lhalf * 16;
  const _Float16* img1 = in + (size_t)b1 * imgStride + lhalf * 16;
  __builtin_prefetch(img0, 0, 1);
  const _Float16* wr0 = wlds + lm * Kstride;
  const _Float16* wr1 = wlds + (lm + 16) * Kstride;

  v8f accA0 = {}, accA1 = {}, accB0 = {}, accB1 = {};
  for (int ks = 0; ks < Ksteps; ++ks) {
    int kb = ks * 32;
    int tap = ks / cpt;
    int icb = (ks - tap * cpt) * 32;
    int dy = tap / 3, dx = tap - (tap / 3) * 3;
    v16h bt0 = *(const v16h*)(img0 + ((y0 + dy) * Wh + (x0 + dx)) * Cinp + icb);
    v16h bt1 = *(const v16h*)(img1 + ((y1 + dy) * Wh + (x1 + dx)) * Cinp + icb);
    v16h a0 = loadA(wr0, kb, lhalf);
    v16h a1 = loadA(wr1, kb, lhalf);
    accA0 = wmma16(a0, bt0, accA0);
    accA1 = wmma16(a1, bt0, accA1);
    accB0 = wmma16(a0, bt1, accB0);
    accB1 = wmma16(a1, bt1, accB1);
  }
  _Float16* ob0 = outpre + ((size_t)(b0 * H + y0) * W + x0) * 32;
  _Float16* ob1 = outpre + ((size_t)(b1 * H + y1) * W + x1) * 32;
  v8h sA0, sA1, sB0, sB1;
#pragma unroll
  for (int r = 0; r < 8; ++r) {
    int m = lhalf * 8 + r;
    sA0[r] = (_Float16)(accA0[r] + bias[m]);
    sA1[r] = (_Float16)(accA1[r] + bias[m + 16]);
    sB0[r] = (_Float16)(accB0[r] + bias[m]);
    sB1[r] = (_Float16)(accB1[r] + bias[m + 16]);
  }
  *(v8h*)(ob0 + lhalf * 8)      = sA0;
  *(v8h*)(ob0 + 16 + lhalf * 8) = sA1;
  *(v8h*)(ob1 + lhalf * 8)      = sB0;
  *(v8h*)(ob1 + 16 + lhalf * 8) = sB1;
}

// ---------------- FC heads ----------------
__global__ __launch_bounds__(256) void fc_wmma_kernel(
    const _Float16* __restrict__ h3, const _Float16* __restrict__ w1,
    const _Float16* __restrict__ w2, const float* __restrict__ b1,
    const float* __restrict__ b2, float* __restrict__ mu, float* __restrict__ logvar) {
  const int lane = threadIdx.x & 31;
  const int wid  = (blockIdx.x * blockDim.x + threadIdx.x) >> 5;
  const int lhalf = lane >> 4, lm = lane & 15;
  int mtile = wid & 31;
  int ntile = wid >> 5;
  int n = ntile * 16 + lm;
  v8f acc1 = {}, acc2 = {};
  const _Float16* r1 = w1 + (size_t)(mtile * 16 + lm) * 512;
  const _Float16* r2 = w2 + (size_t)(mtile * 16 + lm) * 512;
  const _Float16* hb = h3 + (size_t)n * 512 + lhalf * 16;
  for (int ks = 0; ks < 16; ++ks) {
    int kb = ks * 32;
    v16h bt = *(const v16h*)(hb + kb);
    v16h aA = loadA(r1, kb, lhalf);
    v16h aB = loadA(r2, kb, lhalf);
    acc1 = wmma16(aA, bt, acc1);
    acc2 = wmma16(aB, bt, acc2);
  }
  int mb = mtile * 16 + lhalf * 8;
  v4f m0, m1, l0, l1;
#pragma unroll
  for (int r = 0; r < 4; ++r) {
    m0[r] = acc1[r] + b1[mb + r];
    m1[r] = acc1[r + 4] + b1[mb + 4 + r];
    l0[r] = acc2[r] + b2[mb + r];
    l1[r] = acc2[r + 4] + b2[mb + 4 + r];
  }
  *(v4f*)(mu + (size_t)n * 512 + mb)         = m0;
  *(v4f*)(mu + (size_t)n * 512 + mb + 4)     = m1;
  *(v4f*)(logvar + (size_t)n * 512 + mb)     = l0;
  *(v4f*)(logvar + (size_t)n * 512 + mb + 4) = l1;
}

// ---------------- decoder: LDS-staged dilated transposed conv, NHWC ----------------
// LDS tile [IMGS][Dh][Dh][16] (Dh = 2*HIN+4, zero-filled; input (iy,ix) at (2iy+2, 2ix+2)).
template <int HIN, int IMGS, int TPW>
__global__ __launch_bounds__(256) void deconv_wmma(
    const _Float16* __restrict__ in, const _Float16* __restrict__ wpk,
    const float* __restrict__ bias, _Float16* __restrict__ out16,
    float* __restrict__ outF, int finalLayer) {
  constexpr int Dh   = 2 * HIN + 4;        // reads reach row/col oy+ky <= 2*HIN+3
  constexpr int DhDh = Dh * Dh;
  constexpr int HOUT = 2 * HIN;
  constexpr int TPI  = (HOUT * HOUT) / 16;
  static_assert((TPW & 1) == 0, "pairwise N-blocking");
  __shared__ __align__(64) _Float16 tile[IMGS * DhDh * 16];
  __shared__ __align__(64) _Float16 wlds[16 * 416];

  const int tid = threadIdx.x;
  const int lane = tid & 31, wv = tid >> 5, lhalf = lane >> 4, lm = lane & 15;
  const int img0 = blockIdx.x * IMGS;
  const _Float16* src = in + (size_t)img0 * (HIN * HIN * 16);   // NHWC input
  __builtin_prefetch(src, 0, 1);

  for (int i = tid; i < 16 * 416; i += 256) wlds[i] = wpk[i];
  v16h zv = {};
  for (int i = tid; i < IMGS * DhDh; i += 256) *(v16h*)(tile + i * 16) = zv;
  __syncthreads();
  for (int i = tid; i < IMGS * HIN * HIN; i += 256) {
    int il = i / (HIN * HIN);
    int p  = i - il * (HIN * HIN);
    int iy = p / HIN, ix = p - iy * HIN;
    *(v16h*)(tile + ((il * Dh + 2 * iy + 2) * Dh + (2 * ix + 2)) * 16) =
        *(const v16h*)(src + (size_t)i * 16);
  }
  __syncthreads();

  const _Float16* wrow = wlds + lm * 416;
  for (int t = 0; t < TPW; t += 2) {
    int tile0 = wv * TPW + t, tile1 = tile0 + 1;
    int il0 = tile0 / TPI, tp0 = tile0 - il0 * TPI;
    int il1 = tile1 / TPI, tp1 = tile1 - il1 * TPI;
    int pos0 = tp0 * 16 + lm, pos1 = tp1 * 16 + lm;
    int oy0 = pos0 / HOUT, ox0 = pos0 - oy0 * HOUT;
    int oy1 = pos1 / HOUT, ox1 = pos1 - oy1 * HOUT;
    v8f acc0 = {}, acc1 = {};
    for (int ks = 0; ks < 13; ++ks) {
      int kb = ks * 32;
      int tap = ks * 2 + lhalf;
      if (tap > 24) tap = 24;              // chunk-12 upper slots: A rows are zero
      int ky = tap / 5, kx = tap - (tap / 5) * 5;
      v16h bt0 = *(const v16h*)(tile + ((il0 * Dh + oy0 + ky) * Dh + (ox0 + kx)) * 16);
      v16h bt1 = *(const v16h*)(tile + ((il1 * Dh + oy1 + ky) * Dh + (ox1 + kx)) * 16);
      v16h a = loadA(wrow, kb, lhalf);
      acc0 = wmma16(a, bt0, acc0);
      acc1 = wmma16(a, bt1, acc1);
    }
    int imgA = img0 + il0, imgB = img0 + il1;
    if (!finalLayer) {
      v8h s0, s1;
#pragma unroll
      for (int r = 0; r < 8; ++r) {
        int m = lhalf * 8 + r;
        s0[r] = (_Float16)fmaxf(acc0[r] + bias[m], 0.0f);
        s1[r] = (_Float16)fmaxf(acc1[r] + bias[m], 0.0f);
      }
      *(v8h*)(out16 + ((size_t)(imgA * HOUT + oy0) * HOUT + ox0) * 16 + lhalf * 8) = s0;
      *(v8h*)(out16 + ((size_t)(imgB * HOUT + oy1) * HOUT + ox1) * 16 + lhalf * 8) = s1;
    } else {
#pragma unroll
      for (int r = 0; r < 8; ++r) {
        int m = lhalf * 8 + r;
        if (m < 3) {
          float s0 = 1.0f / (1.0f + expf(-(acc0[r] + bias[m])));
          float s1 = 1.0f / (1.0f + expf(-(acc1[r] + bias[m])));
          int tA = imgA >> 8, bA = imgA & 255;
          int tB = imgB >> 8, bB = imgB & 255;
          outF[((size_t)bA * 90 + tA * 3 + m) * 1024 + oy0 * 32 + ox0] = s0;
          outF[((size_t)bB * 90 + tB * 3 + m) * 1024 + oy1 * 32 + ox1] = s1;
        }
      }
    }
  }
}

// ---------------- host launcher ----------------

extern "C" void kernel_launch(void* const* d_in, const int* in_sizes, int n_in,
                              void* d_out, int out_size, void* d_ws, size_t ws_size,
                              hipStream_t stream) {
  (void)in_sizes; (void)n_in; (void)out_size; (void)ws_size;
  const float* x    = (const float*)d_in[0];
  const float* eps  = (const float*)d_in[1];
  const float* c1w  = (const float*)d_in[2];
  const float* c1b  = (const float*)d_in[3];
  const float* c2w  = (const float*)d_in[4];
  const float* c2b  = (const float*)d_in[5];
  const float* c3w  = (const float*)d_in[6];
  const float* c3b  = (const float*)d_in[7];
  const float* fc1w = (const float*)d_in[8];
  const float* fc1b = (const float*)d_in[9];
  const float* fc2w = (const float*)d_in[10];
  const float* fc2b = (const float*)d_in[11];
  const float* hamw = (const float*)d_in[12];
  const float* d1w  = (const float*)d_in[13];
  const float* d1b  = (const float*)d_in[14];
  const float* d2w  = (const float*)d_in[15];
  const float* d2b  = (const float*)d_in[16];
  const float* d3w  = (const float*)d_in[17];
  const float* d3b  = (const float*)d_in[18];

  float* out = (float*)d_out;
  const size_t RESULT_N = (size_t)256 * 90 * 32 * 32;
  float* mu     = out + RESULT_N;
  float* logvar = mu + 256 * 512;

  char* ws = (char*)d_ws;
  // all NHWC; offsets 256-aligned; D2 aliases dead XHP+C1PRE region
  const size_t OFF_XHP   = 0;            // 56,819,712 B  [256][34][34][96] f16
  const size_t OFF_C1PRE = 56819712;     // 16,777,216 B  [256][32][32][32] f16
  const size_t OFF_D2    = 0;            // 62,914,560 B  [7680][16][16][16] (aliases)
  const size_t OFF_H1P   = 73596928;     //  5,308,416 B  [256][18][18][32]
  const size_t OFF_C2PRE = 78905344;     //  4,194,304 B  [256][16][16][32]
  const size_t OFF_H2P   = 83099648;     //  1,638,400 B  [256][10][10][32]
  const size_t OFF_C3PRE = 84738048;     //  1,048,576 B  [256][8][8][32]
  const size_t OFF_H3    = 85786624;     //    262,144 B  [256][512] (NCHW-flat)
  const size_t OFF_QSH   = 86048768;     //  3,932,160 B  [7680][4][4][16]
  const size_t OFF_D1    = 89980928;     // 15,728,640 B  [7680][8][8][16]
  const size_t OFF_W1P   = 105709568;
  const size_t OFF_W2P   = 105764864;
  const size_t OFF_W3P   = 105783296;
  const size_t OFF_FC1H  = 105801728;
  const size_t OFF_FC2H  = 106326016;
  const size_t OFF_D1P   = 106850304;
  const size_t OFF_D2P   = 106863616;
  const size_t OFF_D3P   = 106876928;

  _Float16* xhp   = (_Float16*)(ws + OFF_XHP);
  _Float16* c1pre = (_Float16*)(ws + OFF_C1PRE);
  _Float16* h1p   = (_Float16*)(ws + OFF_H1P);
  _Float16* c2pre = (_Float16*)(ws + OFF_C2PRE);
  _Float16* h2p   = (_Float16*)(ws + OFF_H2P);
  _Float16* c3pre = (_Float16*)(ws + OFF_C3PRE);
  _Float16* h3    = (_Float16*)(ws + OFF_H3);
  _Float16* qsh   = (_Float16*)(ws + OFF_QSH);
  _Float16* d1    = (_Float16*)(ws + OFF_D1);
  _Float16* d2    = (_Float16*)(ws + OFF_D2);
  _Float16* w1p   = (_Float16*)(ws + OFF_W1P);
  _Float16* w2p   = (_Float16*)(ws + OFF_W2P);
  _Float16* w3p   = (_Float16*)(ws + OFF_W3P);
  _Float16* fc1h  = (_Float16*)(ws + OFF_FC1H);
  _Float16* fc2h  = (_Float16*)(ws + OFF_FC2H);
  _Float16* d1p   = (_Float16*)(ws + OFF_D1P);
  _Float16* d2p   = (_Float16*)(ws + OFF_D2P);
  _Float16* d3p   = (_Float16*)(ws + OFF_D3P);

  const dim3 blk(256);

  // conversion + packing
  pad_input_x<<<dim3(110976), blk, 0, stream>>>(x, xhp);
  cvt_f32_to_f16<<<dim3(1024), blk, 0, stream>>>(fc1w, fc1h, 512 * 512);
  cvt_f32_to_f16<<<dim3(1024), blk, 0, stream>>>(fc2w, fc2h, 512 * 512);
  pack_enc_w<<<dim3(108), blk, 0, stream>>>(c1w, w1p, 90, 3, 27);
  pack_enc_w<<<dim3(36),  blk, 0, stream>>>(c2w, w2p, 32, 1, 9);
  pack_enc_w<<<dim3(36),  blk, 0, stream>>>(c3w, w3p, 32, 1, 9);
  pack_dec_w<<<dim3(26),  blk, 0, stream>>>(d1w, d1p, 16);
  pack_dec_w<<<dim3(26),  blk, 0, stream>>>(d2w, d2p, 16);
  pack_dec_w<<<dim3(26),  blk, 0, stream>>>(d3w, d3p, 3);

  // encoder (NHWC)
  conv_enc_wmma<<<dim3(1024), blk, 0, stream>>>(xhp, w1p, c1b, c1pre,
      34, 34, 32, 32, 27, 3, 864, 96);
  maxpool_pad_nhwc<<<dim3(10368), blk, 0, stream>>>(c1pre, h1p, 256, 32, 32, 18, 18, 1);
  conv_enc_wmma<<<dim3(256), blk, 0, stream>>>(h1p, w2p, c2b, c2pre,
      18, 18, 16, 16, 9, 1, 288, 32);
  maxpool_pad_nhwc<<<dim3(3200), blk, 0, stream>>>(c2pre, h2p, 256, 16, 16, 10, 10, 1);
  conv_enc_wmma<<<dim3(64), blk, 0, stream>>>(h2p, w3p, c3b, c3pre,
      10, 10, 8, 8, 9, 1, 288, 32);
  maxpool3_to_h3<<<dim3(512), blk, 0, stream>>>(c3pre, h3);

  // FC heads -> mu, logvar (f32, in d_out)
  fc_wmma_kernel<<<dim3(64), blk, 0, stream>>>(h3, fc1h, fc2h, fc1b, fc2b, mu, logvar);

  // reparameterize + rollout (NHWC qs)
  qs_kernel<<<dim3(7680), blk, 0, stream>>>(mu, logvar, eps, hamw, qsh);

  // decoder
  deconv_wmma<4, 4, 2><<<dim3(1920), blk, 0, stream>>>(qsh, d1p, d1b, d1, nullptr, 0);
  deconv_wmma<8, 1, 2><<<dim3(7680), blk, 0, stream>>>(d1, d2p, d2b, d2, nullptr, 0);
  deconv_wmma<16, 1, 8><<<dim3(7680), blk, 0, stream>>>(d2, d3p, d3b, nullptr, out, 1);
}
